// TransformerEncoderLayer_1743756722932
// MI455X (gfx1250) — matrix operand order repaired and checked by reading, not compiled, not run on hardware
//
#include <hip/hip_runtime.h>
#include <hip/hip_bf16.h>

// ---------------------------------------------------------------------------
// Problem constants (match reference)
// ---------------------------------------------------------------------------
#define S_LEN   2048
#define BATCH   8
#define DMODEL  1024
#define DFF     2048
#define MROWS   (S_LEN * BATCH)     // 16384
#define LN_EPS  1e-5f
#define SM_SCALE 0.1767766952966369f   // 1 / 1024^0.25

typedef __attribute__((ext_vector_type(16))) __bf16 bf16x16;
typedef __attribute__((ext_vector_type(8)))  float  f32x8;
typedef __attribute__((ext_vector_type(4)))  unsigned int u32x4;
typedef __attribute__((ext_vector_type(8)))  int i32x8;
typedef __attribute__((ext_vector_type(4)))  int i32x4;

typedef __hip_bfloat16 bf16_t;

__device__ __forceinline__ unsigned short f32_to_bf16_bits(float f) {
    bf16_t h = __float2bfloat16(f);
    return *reinterpret_cast<unsigned short*>(&h);
}

// ---------------------------------------------------------------------------
// TDM: issue a 2D tile load global->LDS (bf16 elements).
//   Rows of tile_w_elems, separated in memory by row_stride_elems.
//   LDS pad: 1 DWORD every 16 DWORDs (pad_interval=3, pad_amount=0), so a
//   64-element row (32 dw) lands with pitch 34 dw; chunk c at dword c*17.
// Descriptor layout per CDNA5 ISA 8.3/8.4 (D# group0/group1).
// ---------------------------------------------------------------------------
__device__ __forceinline__ void tdm_load_tile_2d(unsigned int lds_off,
                                                 const void* gptr,
                                                 int tile_w_elems, int tile_rows,
                                                 long row_stride_elems) {
    unsigned long long ga = (unsigned long long)(uintptr_t)gptr;
    unsigned long long sb = (unsigned long long)row_stride_elems; // data_size units
    u32x4 g0;
    g0.x = 1u;                                            // count=1, user mode
    g0.y = lds_off;                                       // lds_addr (bytes)
    g0.z = (unsigned int)(ga & 0xffffffffu);              // global_addr[31:0]
    g0.w = (unsigned int)((ga >> 32) & 0x01ffffffu) |     // global_addr[56:32]
           0x80000000u;                                   // type=2 ("image")
    i32x8 g1;
    g1[0] = (1 << 16)      // data_size = 1 -> 2 bytes
          | (1 << 20)      // pad_enable
          | (3 << 22);     // pad_interval = 3 -> every 16 DWORDs; pad_amount=0 -> 1 DWORD
    g1[1] = (tile_w_elems & 0xffff) << 16;                            // tensor_dim0[15:0]
    g1[2] = ((tile_w_elems >> 16) & 0xffff) | ((tile_rows & 0xffff) << 16); // td0 hi | td1 lo
    g1[3] = ((tile_rows >> 16) & 0xffff) | ((tile_w_elems & 0xffff) << 16); // td1 hi | tile_dim0
    g1[4] = (tile_rows & 0xffff);                                     // tile_dim1 (tile_dim2=0)
    g1[5] = (int)(sb & 0xffffffffu);                                  // dim0_stride[31:0]
    g1[6] = (int)((sb >> 32) & 0xffffu);                              // dim0_stride[47:32]
    g1[7] = 0;
    i32x4 z4 = {0, 0, 0, 0};
#if __clang_major__ >= 23
    i32x8 z8 = {0, 0, 0, 0, 0, 0, 0, 0};
    __builtin_amdgcn_tensor_load_to_lds(g0, g1, z4, z4, z8, 0);
#else
    __builtin_amdgcn_tensor_load_to_lds(g0, g1, z4, z4, 0);
#endif
}

// ---------------------------------------------------------------------------
// Elementwise fp32 -> bf16 cast (vectorized, n multiple of 4)
// ---------------------------------------------------------------------------
__global__ __launch_bounds__(256) void cast_f32_to_bf16(const float* __restrict__ src,
                                                        bf16_t* __restrict__ dst, int n) {
    int i4 = blockIdx.x * blockDim.x + threadIdx.x;
    int n4 = n >> 2;
    if (i4 >= n4) return;
    float4 f = reinterpret_cast<const float4*>(src)[i4];
    unsigned int u0 = (unsigned int)f32_to_bf16_bits(f.x) |
                      ((unsigned int)f32_to_bf16_bits(f.y) << 16);
    unsigned int u1 = (unsigned int)f32_to_bf16_bits(f.z) |
                      ((unsigned int)f32_to_bf16_bits(f.w) << 16);
    reinterpret_cast<uint2*>(dst)[i4] = make_uint2(u0, u1);
}

// ---------------------------------------------------------------------------
// Generic batched bf16 GEMM, f32 accumulate, v_wmma_f32_16x16x32_bf16.
//   C[m,n] = sum_k A[m,k]*B(k,n) (+bias[n]) (opt ReLU)
//   bIsKN==0 : Bm is W[N,K] row-major (C = A*W^T)  -- TDM-staged
//   bIsKN==1 : Bm is B[K,N] row-major (C = A*B)    -- manual transpose staging
//   A tile always TDM-staged. Tiles BM=64, BN=256, BK=64 (2 WMMA chunks of 32).
//   Double-buffered LDS: wave 0 issues tile i+1 before computing tile i and
//   waits TENSORcnt<=inflight (per-wave TDM completes in order).
//   Assumes M%64==0, N%256==0, K%64==0 (true for all uses here).
// ---------------------------------------------------------------------------
__global__ __launch_bounds__(256) void gemm_bf16_wmma(
    const bf16_t* __restrict__ A,  long aBatch, int lda,
    const bf16_t* __restrict__ Bm, long bBatch, int ldb, int bIsKN,
    float* __restrict__ outF, bf16_t* __restrict__ outB, long cBatch, int ldc,
    const float* __restrict__ bias, int relu, int K)
{
    // Row pitch 34 DWORDs (64 bf16 + 2 TDM pad DWORDs). Double buffered.
    __shared__ unsigned int Asm[2][64 * 34];     // 17.4 KB : A tiles [64 m][64 k]
    __shared__ unsigned int Bsm[2][256 * 34];    // 69.6 KB : B tiles [256 n][64 k]

    const int tid  = threadIdx.x;
    const int lane = tid & 31;
    const int wid  = tid >> 5;
    const int wm   = wid >> 2;          // 0..1  (M direction)
    const int wn   = wid & 3;           // 0..3  (N direction)
    const int mloc = lane & 15;
    const int hi   = (lane >> 4) & 1;

    const int cm0 = blockIdx.y * 64;
    const int cn0 = blockIdx.x * 256;

    const bf16_t* Ab = A  + (long)blockIdx.z * aBatch;
    const bf16_t* Bb = Bm + (long)blockIdx.z * bBatch;

    f32x8 acc[2][4];
#pragma unroll
    for (int mi = 0; mi < 2; ++mi)
#pragma unroll
        for (int ni = 0; ni < 4; ++ni)
#pragma unroll
            for (int e = 0; e < 8; ++e) acc[mi][ni][e] = 0.0f;

    union Frag { bf16x16 v; unsigned int u[8]; };

    const int steps = K >> 6;   // K-tiles of 64

    // prologue: stage tile 0 into buffer 0
    if (wid == 0) {
        tdm_load_tile_2d((unsigned int)(uintptr_t)&Asm[0][0],
                         Ab + (long)cm0 * lda, 64, 64, lda);
        if (!bIsKN)
            tdm_load_tile_2d((unsigned int)(uintptr_t)&Bsm[0][0],
                             Bb + (long)cn0 * ldb, 64, 256, ldb);
    }

    for (int i = 0; i < steps; ++i) {
        const int cur = i & 1;
        const int k0  = i << 6;

        if (bIsKN) {
            // B[K,N] row-major: read 2 n's per uint, transpose into [n][k] in LDS
            unsigned short* Bs = reinterpret_cast<unsigned short*>(&Bsm[cur][0]); // pitch 68
#pragma unroll
            for (int r = 0; r < 32; ++r) {
                int idx = tid + r * 256;           // 8192 uints = 64 k x 128 n-pairs
                int k  = idx >> 7;                  // 0..63
                int n2 = idx & 127;                 // n-pair 0..127
                unsigned int u = *reinterpret_cast<const unsigned int*>(
                    Bb + (long)(k0 + k) * ldb + cn0 + 2 * n2);
                int pos = (k >> 5) * 34 + (k & 31); // chunk offset + within-chunk k
                Bs[(2 * n2 + 0) * 68 + pos] = (unsigned short)(u & 0xffffu);
                Bs[(2 * n2 + 1) * 68 + pos] = (unsigned short)(u >> 16);
            }
            if (i + 1 < steps)   // warm L2 for the next K-block (global_prefetch_b8)
                __builtin_prefetch(Bb + (long)(k0 + 64 + (tid & 63)) * ldb + cn0 + (tid >> 6) * 64,
                                   0, 1);
        }

        if (wid == 0) {
            if (i + 1 < steps) {
                const int nb = cur ^ 1;
                tdm_load_tile_2d((unsigned int)(uintptr_t)&Asm[nb][0],
                                 Ab + (long)cm0 * lda + (k0 + 64), 64, 64, lda);
                if (!bIsKN) {
                    tdm_load_tile_2d((unsigned int)(uintptr_t)&Bsm[nb][0],
                                     Bb + (long)cn0 * ldb + (k0 + 64), 64, 256, ldb);
                    __builtin_amdgcn_s_wait_tensorcnt(2);  // tile i resident, i+1 in flight
                } else {
                    __builtin_amdgcn_s_wait_tensorcnt(1);
                }
            } else {
                __builtin_amdgcn_s_wait_tensorcnt(0);
            }
        }
        __syncthreads();

        // ---- two 32-deep WMMA chunks from the resident 64-deep tile ----
#pragma unroll
        for (int c = 0; c < 2; ++c) {
            const int cb = c * 17;   // chunk base (DWORDs) within a 34-dw row
            Frag af[2];
#pragma unroll
            for (int mi = 0; mi < 2; ++mi)
#pragma unroll
                for (int j = 0; j < 8; ++j)
                    af[mi].u[j] = Asm[cur][(wm * 32 + mi * 16 + mloc) * 34 + cb +
                                           ((j >> 2) * 8 + hi * 4 + (j & 3))];
            Frag bfg[4];
#pragma unroll
            for (int ni = 0; ni < 4; ++ni)
#pragma unroll
                for (int j = 0; j < 8; ++j)
                    bfg[ni].u[j] = Bsm[cur][(wn * 64 + ni * 16 + mloc) * 34 + cb +
                                            ((j >> 2) * 8 + hi * 4 + (j & 3))];
#pragma unroll
            for (int mi = 0; mi < 2; ++mi)
#pragma unroll
                for (int ni = 0; ni < 4; ++ni)
                    acc[mi][ni] = __builtin_amdgcn_wmma_f32_16x16x32_bf16(
                        false, af[mi].v, false, bfg[ni].v,
                        (short)0, acc[mi][ni], false, false);
        }
        __syncthreads();
    }

    // ---- epilogue ----
    float* outFb = outF ? outF + (long)blockIdx.z * cBatch : nullptr;
    bf16_t* outBb = outB ? outB + (long)blockIdx.z * cBatch : nullptr;
#pragma unroll
    for (int mi = 0; mi < 2; ++mi)
#pragma unroll
        for (int ni = 0; ni < 4; ++ni) {
            int col = cn0 + wn * 64 + ni * 16 + mloc;
            float bv = bias ? bias[col] : 0.0f;
#pragma unroll
            for (int e = 0; e < 8; ++e) {
                int row = cm0 + wm * 32 + mi * 16 + e + 8 * hi;
                float v = acc[mi][ni][e] + bv;
                if (relu) v = fmaxf(v, 0.0f);
                long off = (long)row * ldc + col;
                if (outFb) outFb[off] = v;
                if (outBb) outBb[off] = __float2bfloat16(v);
            }
        }
}

// ---------------------------------------------------------------------------
// Row softmax over bf16 scores [B, S, S], in place.
// scale d^-1/4 then mask_pad[k,b] => -inf, then softmax.
// One wave per row of S=2048; 8 rows per 256-thread block.
// ---------------------------------------------------------------------------
__global__ __launch_bounds__(256) void softmax_rows(bf16_t* __restrict__ sc,
                                                    const unsigned char* __restrict__ mask) {
    const int lane = threadIdx.x & 31;
    const int wid  = threadIdx.x >> 5;
    const int rid  = blockIdx.x * 8 + wid;          // rid = b*S + q
    const int b    = rid / S_LEN;

    unsigned int* rp = reinterpret_cast<unsigned int*>(sc + (long)rid * S_LEN);

    float v[64];
    float mx = -__builtin_inff();
#pragma unroll
    for (int j = 0; j < 32; ++j) {
        unsigned int u = rp[lane + j * 32];
        int k0 = (lane + j * 32) * 2;
        float f0 = __uint_as_float(u << 16) * SM_SCALE;
        float f1 = __uint_as_float(u & 0xffff0000u) * SM_SCALE;
        if (mask[(long)k0 * BATCH + b])       f0 = -__builtin_inff();
        if (mask[(long)(k0 + 1) * BATCH + b]) f1 = -__builtin_inff();
        v[2 * j] = f0; v[2 * j + 1] = f1;
        mx = fmaxf(mx, fmaxf(f0, f1));
    }
#pragma unroll
    for (int off = 16; off >= 1; off >>= 1)
        mx = fmaxf(mx, __shfl_xor(mx, off, 32));

    float s = 0.0f;
#pragma unroll
    for (int e = 0; e < 64; ++e) { float ex = __expf(v[e] - mx); v[e] = ex; s += ex; }
#pragma unroll
    for (int off = 16; off >= 1; off >>= 1)
        s += __shfl_xor(s, off, 32);
    float inv = 1.0f / s;

#pragma unroll
    for (int j = 0; j < 32; ++j) {
        unsigned int u = (unsigned int)f32_to_bf16_bits(v[2 * j] * inv) |
                         ((unsigned int)f32_to_bf16_bits(v[2 * j + 1] * inv) << 16);
        rp[lane + j * 32] = u;
    }
}

// ---------------------------------------------------------------------------
// Fused residual add + LayerNorm over D=1024; block per row, 4 elems/thread.
//   outF = LN(a + b) * g + be (f32); optional bf16 copy in outB.
// ---------------------------------------------------------------------------
__global__ __launch_bounds__(256) void residual_layernorm(
    const float* __restrict__ a, const float* __restrict__ b,
    const float* __restrict__ g, const float* __restrict__ be,
    float* __restrict__ outF, bf16_t* __restrict__ outB)
{
    __shared__ float sS[8], sQ[8], stat[2];
    const int tid = threadIdx.x;
    const int lane = tid & 31, wid = tid >> 5;
    const long base = (long)blockIdx.x * DMODEL;
    const int e0 = tid * 4;

    float4 av = *reinterpret_cast<const float4*>(a + base + e0);
    float4 bv = *reinterpret_cast<const float4*>(b + base + e0);
    float r0 = av.x + bv.x, r1 = av.y + bv.y, r2 = av.z + bv.z, r3 = av.w + bv.w;

    float sum = r0 + r1 + r2 + r3;
    float sq  = r0 * r0 + r1 * r1 + r2 * r2 + r3 * r3;
#pragma unroll
    for (int off = 16; off >= 1; off >>= 1) {
        sum += __shfl_xor(sum, off, 32);
        sq  += __shfl_xor(sq,  off, 32);
    }
    if (lane == 0) { sS[wid] = sum; sQ[wid] = sq; }
    __syncthreads();
    if (tid == 0) {
        float ts = 0.0f, tq = 0.0f;
#pragma unroll
        for (int w = 0; w < 8; ++w) { ts += sS[w]; tq += sQ[w]; }
        float mean = ts * (1.0f / DMODEL);
        float var  = tq * (1.0f / DMODEL) - mean * mean;
        stat[0] = mean;
        stat[1] = rsqrtf(var + LN_EPS);
    }
    __syncthreads();
    float mean = stat[0], rst = stat[1];

    float rr[4] = {r0, r1, r2, r3};
#pragma unroll
    for (int c = 0; c < 4; ++c) {
        float val = (rr[c] - mean) * rst * g[e0 + c] + be[e0 + c];
        outF[base + e0 + c] = val;
        if (outB) outB[base + e0 + c] = __float2bfloat16(val);
    }
}

// ---------------------------------------------------------------------------
// Host launcher
// ---------------------------------------------------------------------------
static inline void launch_gemm(hipStream_t stream, int M, int N, int batches,
                               const bf16_t* A, long aB, int lda,
                               const bf16_t* Bm, long bB, int ldb, int bIsKN,
                               float* outF, bf16_t* outB, long cB, int ldc,
                               const float* bias, int relu, int K) {
    dim3 grid(N / 256, M / 64, batches);
    gemm_bf16_wmma<<<grid, 256, 0, stream>>>(A, aB, lda, Bm, bB, ldb, bIsKN,
                                             outF, outB, cB, ldc, bias, relu, K);
}

extern "C" void kernel_launch(void* const* d_in, const int* in_sizes, int n_in,
                              void* d_out, int out_size, void* d_ws, size_t ws_size,
                              hipStream_t stream) {
    (void)in_sizes; (void)n_in; (void)out_size; (void)ws_size;

    const float* x    = (const float*)d_in[0];
    const unsigned char* mask = (const unsigned char*)d_in[1];
    const float* Wq   = (const float*)d_in[2];
    const float* Wk   = (const float*)d_in[3];
    const float* Wv   = (const float*)d_in[4];
    const float* Wo   = (const float*)d_in[5];
    const float* bo   = (const float*)d_in[6];
    const float* W1   = (const float*)d_in[7];
    const float* b1   = (const float*)d_in[8];
    const float* W2   = (const float*)d_in[9];
    const float* b2   = (const float*)d_in[10];
    const float* g1   = (const float*)d_in[11];
    const float* be1  = (const float*)d_in[12];
    const float* g2   = (const float*)d_in[13];
    const float* be2  = (const float*)d_in[14];
    float* out = (float*)d_out;

    // ---- workspace layout; aliases reuse dead regions ----
    char* ws = (char*)d_ws;
    const size_t MB32 = (size_t)MROWS * DMODEL * sizeof(bf16_t);   // 32 MB
    bf16_t* xb  = (bf16_t*)(ws);
    bf16_t* qb  = (bf16_t*)(ws + MB32);              // later reused as attn
    bf16_t* kb  = (bf16_t*)(ws + 2 * MB32);          // kb+vb later reused as zf
    bf16_t* vb  = (bf16_t*)(ws + 3 * MB32);
    bf16_t* sc  = (bf16_t*)(ws + 4 * MB32);          // [B,S,S] bf16 = 64 MB; later hb
    bf16_t* x1b = (bf16_t*)(ws + 4 * MB32 + (size_t)BATCH * S_LEN * S_LEN * sizeof(bf16_t));
    char*  wsc  = (char*)x1b + MB32;
    bf16_t* wqb = (bf16_t*)(wsc);
    bf16_t* wkb = (bf16_t*)(wsc + 2u  * 1024 * 1024);
    bf16_t* wvb = (bf16_t*)(wsc + 4u  * 1024 * 1024);
    bf16_t* wob = (bf16_t*)(wsc + 6u  * 1024 * 1024);
    bf16_t* w1b = (bf16_t*)(wsc + 8u  * 1024 * 1024);
    bf16_t* w2b = (bf16_t*)(wsc + 12u * 1024 * 1024);
    bf16_t* attnb = qb;                 // alias: q dead after scores
    bf16_t* hb    = sc;                 // alias: scores dead after attn*V
    float*  zf    = (float*)kb;         // alias: k,v dead after attn*V (64 MB f32)

    // ---- 1) casts to bf16 ----
    auto cast = [&](const float* s, bf16_t* d, int n) {
        cast_f32_to_bf16<<<dim3((n / 4 + 255) / 256), 256, 0, stream>>>(s, d, n);
    };
    cast(x,  xb,  MROWS * DMODEL);
    cast(Wq, wqb, DMODEL * DMODEL);
    cast(Wk, wkb, DMODEL * DMODEL);
    cast(Wv, wvb, DMODEL * DMODEL);
    cast(Wo, wob, DMODEL * DMODEL);
    cast(W1, w1b, DFF * DMODEL);
    cast(W2, w2b, DMODEL * DFF);

    // ---- 2) Q, K, V projections ----
    launch_gemm(stream, MROWS, DMODEL, 1, xb, 0, DMODEL, wqb, 0, DMODEL, 0,
                nullptr, qb, 0, DMODEL, nullptr, 0, DMODEL);
    launch_gemm(stream, MROWS, DMODEL, 1, xb, 0, DMODEL, wkb, 0, DMODEL, 0,
                nullptr, kb, 0, DMODEL, nullptr, 0, DMODEL);
    launch_gemm(stream, MROWS, DMODEL, 1, xb, 0, DMODEL, wvb, 0, DMODEL, 0,
                nullptr, vb, 0, DMODEL, nullptr, 0, DMODEL);

    // ---- 3) scores[b] = Q_b K_b^T (batched; rows strided by B*D) ----
    launch_gemm(stream, S_LEN, S_LEN, BATCH,
                qb, DMODEL, BATCH * DMODEL,
                kb, DMODEL, BATCH * DMODEL, 0,
                nullptr, sc, (long)S_LEN * S_LEN, S_LEN, nullptr, 0, DMODEL);

    // ---- 4) softmax rows (scale + mask), in place ----
    softmax_rows<<<dim3(BATCH * S_LEN / 8), 256, 0, stream>>>(sc, mask);

    // ---- 5) attn[b] = softmax * V_b (B is K x N row-major) ----
    launch_gemm(stream, S_LEN, DMODEL, BATCH,
                sc, (long)S_LEN * S_LEN, S_LEN,
                vb, DMODEL, BATCH * DMODEL, 1,
                nullptr, attnb, DMODEL, BATCH * DMODEL, nullptr, 0, S_LEN);

    // ---- 6) output projection + bo -> f32 into d_out ----
    launch_gemm(stream, MROWS, DMODEL, 1, attnb, 0, DMODEL, wob, 0, DMODEL, 0,
                out, nullptr, 0, DMODEL, bo, 0, DMODEL);

    // ---- 7) LN1: d_out = LN(x + d_out); bf16 copy to x1b ----
    residual_layernorm<<<dim3(MROWS), 256, 0, stream>>>(x, out, g1, be1, out, x1b);

    // ---- 8) FFN1: h = relu(x1 W1^T + b1) -> bf16 ----
    launch_gemm(stream, MROWS, DFF, 1, x1b, 0, DMODEL, w1b, 0, DMODEL, 0,
                nullptr, hb, 0, DFF, b1, 1, DMODEL);

    // ---- 9) FFN2: z = h W2^T + b2 -> f32 ----
    launch_gemm(stream, MROWS, DMODEL, 1, hb, 0, DFF, w2b, 0, DFF, 0,
                zf, nullptr, 0, DMODEL, b2, 0, DFF);

    // ---- 10) LN2: d_out = LN(d_out + z) ----
    residual_layernorm<<<dim3(MROWS), 256, 0, stream>>>(out, zf, g2, be2, out, nullptr);
}
